// ChannelAttention_11819749998720
// MI455X (gfx1250) — matrix operand order repaired and checked by reading, not compiled, not run on hardware
//
#include <hip/hip_runtime.h>
#include <cstdint>
#include <cstddef>

typedef __attribute__((ext_vector_type(16))) __bf16 v16bf;
typedef __attribute__((ext_vector_type(8)))  __bf16 v8bf;
typedef __attribute__((ext_vector_type(8)))  float  v8f;

#define WMMA_BF16(a, b, c) \
  __builtin_amdgcn_wmma_f32_16x16x32_bf16(false, (a), false, (b), (short)0, (c), false, false)

// CDNA5 async global->LDS copy (per-lane 16B), tracked by ASYNCcnt.
static __device__ inline void async_b128(const void* gptr, unsigned lds_addr) {
  asm volatile("global_load_async_to_lds_b128 %0, %1, off"
               :: "v"(lds_addr), "v"(gptr)
               : "memory");
}
static __device__ inline void wait_async0() {
  asm volatile("s_wait_asynccnt 0x0" ::: "memory");
}
static __device__ inline unsigned lds_off(const void* p) {
  return (unsigned)(size_t)p;  // generic LDS ptr: low 32 bits = LDS byte offset
}

// A-fragment (16x32 bf16, M x K): lane<16 holds K 0..15, lane>=16 holds
// K 16..31 of row (lane&15). base = &row[lhalf*16]; second chunk +8 elems.
static __device__ inline v16bf load_frag_a(const __bf16* base) {
  v16bf a;
  *(v8bf*)&a       = *(const v8bf*)(base);
  *((v8bf*)&a + 1) = *(const v8bf*)(base + 8);
  return a;
}
// B-fragment (32x16 bf16, K x N) from LDS stored [n][k]: lane<16 holds K 0..15,
// lane>=16 holds K 16..31 of column (lane&15). base = &col[lhalf*16].
static __device__ inline v16bf load_frag_b(const __bf16* base) {
  v16bf b;
  *(v8bf*)&b       = *(const v8bf*)(base);
  *((v8bf*)&b + 1) = *(const v8bf*)(base + 8);
  return b;
}

// ---------------------------------------------------------------------------
// f32 -> bf16 conversion; weights are transposed to [n][k] so GEMM B-tiles are
// contiguous (async-to-LDS friendly, no in-kernel transpose scatter).
// ---------------------------------------------------------------------------
__global__ void __launch_bounds__(256)
k_convert(const float* __restrict__ x, const float* __restrict__ w,
          const float* __restrict__ pw,
          __bf16* __restrict__ xb, __bf16* __restrict__ wbT,
          __bf16* __restrict__ pwbT, int nx) {
  const int stride = gridDim.x * blockDim.x;
  const int tid0 = blockIdx.x * blockDim.x + threadIdx.x;
  for (int i = tid0; i < nx; i += stride) xb[i] = (__bf16)x[i];
  for (int i = tid0; i < 786432; i += stride) {           // w: 512 x 1536
    const int k = i / 1536, n = i - k * 1536;
    wbT[(size_t)n * 512 + k] = (__bf16)w[i];
  }
  for (int i = tid0; i < 262144; i += stride) {           // pw: 512 x 512
    const int k = i >> 9, n = i & 511;
    pwbT[(size_t)n * 512 + k] = (__bf16)pw[i];
  }
}

// ---------------------------------------------------------------------------
// Tiled bf16 WMMA GEMM: C[M x N] = A[M x K] * Bt^T + bias  (Bt is [n][k])
// Block tile 128x128, K-step 32, 8 waves (wave = 64M x 32N region).
// Double-buffered LDS, async global->LDS staging, one barrier per K-tile.
// mode 0: qkv GEMM (flat-column concat bias, bf16 out); mode 1: proj (f32 out).
// ---------------------------------------------------------------------------
__global__ void __launch_bounds__(256)
k_gemm_bf16(const __bf16* __restrict__ A, const __bf16* __restrict__ Bt,
            __bf16* __restrict__ outb, float* __restrict__ outf,
            const float* __restrict__ biasq, const float* __restrict__ biasv,
            const float* __restrict__ biasp, int N, int K, int mode) {
  __shared__ alignas(16) __bf16 Alds[2][128 * 40];  // [row][k], 16B-aligned rows
  __shared__ alignas(16) __bf16 Blds[2][128 * 40];  // [n][k]

  const int tid   = threadIdx.x;
  const int lane  = tid & 31;
  const int w     = tid >> 5;
  const int l15   = lane & 15;
  const int lhalf = lane >> 4;
  const int wm    = (w >> 2) * 64;
  const int wn    = (w & 3) * 32;
  const size_t bm = (size_t)blockIdx.y * 128;
  const int    bn = blockIdx.x * 128;

  // Stage one 128x32 K-tile of A and of Bt into LDS buffer `buf` (async).
  auto stage = [&](int k0, int buf) {
#pragma unroll
    for (int it = 0; it < 2; ++it) {                 // 512 16B-chunks each
      const int c = tid + (it << 8);
      const int row = c >> 2, kc = (c & 3) << 3;
      async_b128(&A[(bm + row) * K + k0 + kc],
                 lds_off(&Alds[buf][row * 40 + kc]));
      async_b128(&Bt[((size_t)bn + row) * K + k0 + kc],
                 lds_off(&Blds[buf][row * 40 + kc]));
    }
  };

  v8f acc[4][2] = {};
  const int nk = K >> 5;

  stage(0, 0);
  wait_async0();
  __syncthreads();

  for (int t = 0; t < nk; ++t) {
    if (t + 1 < nk) stage((t + 1) << 5, (t + 1) & 1);
    if (t + 2 < nk)  // keep a long-range L2 prefetch in flight
      __builtin_prefetch(&A[(bm + (tid >> 1)) * K + ((t + 2) << 5)], 0, 1);

    const int buf = t & 1;
    v16bf bf0 = load_frag_b(&Blds[buf][(wn + l15) * 40 + lhalf * 16]);
    v16bf bf1 = load_frag_b(&Blds[buf][(wn + 16 + l15) * 40 + lhalf * 16]);
#pragma unroll
    for (int mt = 0; mt < 4; ++mt) {
      v16bf af = load_frag_a(&Alds[buf][(wm + mt * 16 + l15) * 40 + lhalf * 16]);
      acc[mt][0] = WMMA_BF16(af, bf0, acc[mt][0]);
      acc[mt][1] = WMMA_BF16(af, bf1, acc[mt][1]);
    }
    wait_async0();     // next tile landed in the other buffer
    __syncthreads();   // all waves done reading buf -> safe to overwrite next iter
  }

  // epilogue: D layout VGPR r -> (M = r + 8*lhalf, N = l15) within 16x16 tile
#pragma unroll
  for (int mt = 0; mt < 4; ++mt) {
#pragma unroll
    for (int nt = 0; nt < 2; ++nt) {
      const int cg = bn + wn + nt * 16 + l15;
      float bias;
      if (mode == 0)
        bias = (cg < 512) ? biasq[cg] : (cg >= 1024 ? biasv[cg - 1024] : 0.0f);
      else
        bias = biasp[cg];
#pragma unroll
      for (int r = 0; r < 8; ++r) {
        const size_t row = bm + wm + mt * 16 + lhalf * 8 + r;
        const float v = acc[mt][nt][r] + bias;
        if (mode == 0) outb[row * (size_t)N + cg] = (__bf16)v;
        else           outf[row * (size_t)N + cg] = v;
      }
    }
  }
}

// ---------------------------------------------------------------------------
// Per-(b,h,{q|k},channel) inverse L2 norm over the 4096 tokens.
// inv[b*1024 + h*128 + j*64 + dc]
// ---------------------------------------------------------------------------
__global__ void __launch_bounds__(256)
k_sumsq(const __bf16* __restrict__ qkv, float* __restrict__ inv) {
  __shared__ float red[256];
  const int idx = blockIdx.x;
  const int dc = idx & 63, j = (idx >> 6) & 1, h = (idx >> 7) & 7, b = idx >> 10;
  const int col = h * 192 + j * 64 + dc;
  const size_t bOff = (size_t)b * 4096;
  float s = 0.0f;
  for (int n = threadIdx.x; n < 4096; n += 256) {
    const float v = (float)qkv[(bOff + n) * 1536 + col];
    s += v * v;
  }
  red[threadIdx.x] = s;
  __syncthreads();
  for (int off = 128; off > 0; off >>= 1) {
    if (threadIdx.x < off) red[threadIdx.x] += red[threadIdx.x + off];
    __syncthreads();
  }
  if (threadIdx.x == 0) inv[idx] = rsqrtf(fmaxf(red[0], 1.55e-05f));
}

// ---------------------------------------------------------------------------
// Fused channel attention, one block per (b,h):
//   G[64x64] = Q^T K over 4096 tokens (WMMA, channel-major LDS staging)
//   logits = exp(scale)*invq[d]*invk[e]*G ; softmax rows
//   out[tok][d] = sum_e V[tok][e] * attn[d][e] (WMMA) -> bf16 token-major
// ---------------------------------------------------------------------------
struct AttnP1 { __bf16 Qt[64 * 136]; __bf16 Kt[64 * 136]; };  // [ch][tok]
struct AttnP2 { __bf16 Pb[64 * 72];  __bf16 Vs[128 * 72]; };  // [d][e], [tok][e]
union AttnSh  { AttnP1 p1; AttnP2 p2; };

__global__ void __launch_bounds__(256)
k_attn(const __bf16* __restrict__ qkv, const float* __restrict__ inv,
       const float* __restrict__ scale, __bf16* __restrict__ aout) {
  __shared__ alignas(16) AttnSh sh;
  __shared__ float G[64 * 64];
  __shared__ float qn[64], kn[64];

  const int b = blockIdx.x >> 3, h = blockIdx.x & 7;
  const size_t bOff = (size_t)b * 4096;
  const int hOff = h * 192;
  const int tid = threadIdx.x;
  const int lane = tid & 31, w = tid >> 5;
  const int l15 = lane & 15, lhalf = lane >> 4;

  if (tid < 64) {
    qn[tid] = inv[b * 1024 + h * 128 + tid];
    kn[tid] = inv[b * 1024 + h * 128 + 64 + tid];
  }

  // ---- phase 1: Gram matrix G = Q^T K over 32 chunks of 128 tokens
  v8f gacc[2] = {};
  for (int chunk = 0; chunk < 32; ++chunk) {
    const int t0 = chunk << 7;
#pragma unroll
    for (int it = 0; it < 4; ++it) {                 // 1024 8-elem chunks
      const int e = tid + (it << 8);
      const int tok = e >> 3, c8 = (e & 7) << 3;
      const size_t base = (bOff + t0 + tok) * 1536 + hOff;
      uint4 qraw = *(const uint4*)&qkv[base + c8];
      uint4 kraw = *(const uint4*)&qkv[base + 64 + c8];
      const __bf16* qp = (const __bf16*)&qraw;
      const __bf16* kp = (const __bf16*)&kraw;
#pragma unroll
      for (int i = 0; i < 8; ++i) {                  // channel-major transpose
        sh.p1.Qt[(c8 + i) * 136 + tok] = qp[i];
        sh.p1.Kt[(c8 + i) * 136 + tok] = kp[i];
      }
    }
    __syncthreads();
#pragma unroll
    for (int p = 0; p < 2; ++p) {
      const int id = w * 2 + p, i = id >> 2, j = id & 3;  // 16 tiles / 8 waves
#pragma unroll
      for (int kk = 0; kk < 4; ++kk) {
        v16bf af = load_frag_a(&sh.p1.Qt[(i * 16 + l15) * 136 + kk * 32 + lhalf * 16]);
        v16bf bf = load_frag_b(&sh.p1.Kt[(j * 16 + l15) * 136 + kk * 32 + lhalf * 16]);
        gacc[p] = WMMA_BF16(af, bf, gacc[p]);
      }
    }
    __syncthreads();
  }
#pragma unroll
  for (int p = 0; p < 2; ++p) {
    const int id = w * 2 + p, i = id >> 2, j = id & 3;
#pragma unroll
    for (int r = 0; r < 8; ++r)
      G[(i * 16 + lhalf * 8 + r) * 64 + j * 16 + l15] = gacc[p][r];
  }
  __syncthreads();

  // ---- softmax over e for each of the 64 d-rows
  if (tid < 64) {
    const float es = __expf(scale[h]) * qn[tid];
    float mx = -1e30f;
    for (int c = 0; c < 64; ++c) mx = fmaxf(mx, G[tid * 64 + c] * es * kn[c]);
    float s = 0.0f;
    for (int c = 0; c < 64; ++c) {
      const float e = __expf(G[tid * 64 + c] * es * kn[c] - mx);
      s += e;
      G[tid * 64 + c] = e;
    }
    const float rs = 1.0f / s;
    for (int c = 0; c < 64; ++c)
      sh.p2.Pb[tid * 72 + c] = (__bf16)(G[tid * 64 + c] * rs);
  }
  __syncthreads();

  // ---- phase 2: out_chunk[128 x 64] = V_chunk[128 x 64] * attn^T
  for (int chunk = 0; chunk < 32; ++chunk) {
    const int t0 = chunk << 7;
#pragma unroll
    for (int it = 0; it < 4; ++it) {                 // async-stage V chunk
      const int e = tid + (it << 8);
      const int tok = e >> 3, c8 = (e & 7) << 3;
      async_b128(&qkv[(bOff + t0 + tok) * 1536 + hOff + 128 + c8],
                 lds_off(&sh.p2.Vs[tok * 72 + c8]));
    }
    wait_async0();
    __syncthreads();
#pragma unroll
    for (int q = 0; q < 4; ++q) {
      const int id = w * 4 + q, mt = id >> 2, nt = id & 3;  // 32 tiles / 8 waves
      v8f oacc = {};
#pragma unroll
      for (int kk = 0; kk < 2; ++kk) {
        v16bf af = load_frag_a(&sh.p2.Vs[(mt * 16 + l15) * 72 + kk * 32 + lhalf * 16]);
        v16bf bf = load_frag_b(&sh.p2.Pb[(nt * 16 + l15) * 72 + kk * 32 + lhalf * 16]);
        oacc = WMMA_BF16(af, bf, oacc);
      }
#pragma unroll
      for (int r = 0; r < 8; ++r) {
        const size_t row = bOff + t0 + mt * 16 + lhalf * 8 + r;
        aout[row * 512 + h * 64 + nt * 16 + l15] = (__bf16)oacc[r];
      }
    }
    __syncthreads();
  }
}

// ---------------------------------------------------------------------------
extern "C" void kernel_launch(void* const* d_in, const int* in_sizes, int n_in,
                              void* d_out, int out_size, void* d_ws, size_t ws_size,
                              hipStream_t stream) {
  const float* x      = (const float*)d_in[0];  // 16*64*64*512
  const float* qkv_w  = (const float*)d_in[1];  // 512*1536
  const float* q_bias = (const float*)d_in[2];  // 512
  const float* v_bias = (const float*)d_in[3];  // 512
  const float* scale  = (const float*)d_in[4];  // 8
  const float* proj_w = (const float*)d_in[5];  // 512*512
  const float* proj_b = (const float*)d_in[6];  // 512
  float* out = (float*)d_out;

  char* ws = (char*)d_ws;
  size_t off = 0;
  __bf16* xb    = (__bf16*)(ws + off); off += (size_t)33554432 * 2;      // x bf16
  __bf16* wbT   = (__bf16*)(ws + off); off += (size_t)786432 * 2;        // qkv_w^T bf16
  __bf16* pwbT  = (__bf16*)(ws + off); off += (size_t)262144 * 2;        // proj_w^T bf16
  __bf16* qkvb  = (__bf16*)(ws + off); off += (size_t)65536 * 1536 * 2;  // qkv bf16
  float*  inv   = (float*) (ws + off); off += (size_t)16384 * 4;         // inv norms
  __bf16* ab    = (__bf16*)(ws + off); off += (size_t)65536 * 512 * 2;   // attn out bf16

  k_convert<<<8192, 256, 0, stream>>>(x, qkv_w, proj_w, xb, wbT, pwbT, 33554432);
  // qkv = x @ qkv_w + bias : M=65536, N=1536, K=512
  k_gemm_bf16<<<dim3(12, 512), 256, 0, stream>>>(xb, wbT, qkvb, nullptr,
                                                 q_bias, v_bias, nullptr,
                                                 1536, 512, 0);
  k_sumsq<<<16384, 256, 0, stream>>>(qkvb, inv);
  k_attn<<<128, 256, 0, stream>>>(qkvb, inv, scale, ab);
  // out = attn_out @ proj_w + proj_b : M=65536, N=512, K=512
  k_gemm_bf16<<<dim3(4, 512), 256, 0, stream>>>(ab, pwbT, nullptr, out,
                                                nullptr, nullptr, proj_b,
                                                512, 512, 1);
}
